// MultiHeadAttenionBlock_19146964205985
// MI455X (gfx1250) — matrix operand order repaired and checked by reading, not compiled, not run on hardware
//
#include <hip/hip_runtime.h>
#include <hip/hip_bf16.h>

typedef __attribute__((ext_vector_type(16))) _Float16 v16h;
typedef __attribute__((ext_vector_type(8)))  _Float16 v8h;
typedef __attribute__((ext_vector_type(4)))  _Float16 v4h;
typedef __attribute__((ext_vector_type(8)))  float    v8f;
typedef __attribute__((ext_vector_type(4)))  float    v4f;

#define PITCH 72  // f16 LDS row pitch (64 data + 8 pad; 144B rows keep 16B alignment)

__device__ __forceinline__ v16h cat8(v8h lo, v8h hi) {
  return __builtin_shufflevector(lo, hi, 0,1,2,3,4,5,6,7,8,9,10,11,12,13,14,15);
}

// A-fragment: 16x32 f16 tile from row-major LDS matrix X (pitch PITCH).
// Lane L(0-15) holds row L, K = {kc+half*16..+15}  (half = L>>4)
__device__ __forceinline__ v16h load_a(const _Float16* X, int mr, int kc, int lane) {
  int row  = mr + (lane & 15);
  int half = lane >> 4;
  v8h lo = *(const v8h*)&X[row * PITCH + kc + half * 16];
  v8h hi = *(const v8h*)&X[row * PITCH + kc + half * 16 + 8];
  return cat8(lo, hi);
}

// B-fragment: 32x16 f16 tile. BT holds B transposed ([n][k], pitch PITCH), so the
// lane (col n = lane&15, K-half = lane>>4) reads 16 contiguous f16.
__device__ __forceinline__ v16h load_bt(const _Float16* BT, int nb, int kc, int lane) {
  int n  = nb + (lane & 15);
  int kh = lane >> 4;
  v8h lo = *(const v8h*)&BT[n * PITCH + kc + kh * 16];
  v8h hi = *(const v8h*)&BT[n * PITCH + kc + kh * 16 + 8];
  return cat8(lo, hi);
}

// One 16x16 f32 tile of a 64-K GEMM: all four fragments loaded up front so the
// scheduler can overlap ds_load latency with the first wmma.
__device__ __forceinline__ v8f gemm64_tile(const _Float16* A, const _Float16* BT,
                                           int mt, int nt, int lane) {
  v16h a0 = load_a(A, mt, 0, lane);
  v16h b0 = load_bt(BT, nt, 0, lane);
  v16h a1 = load_a(A, mt, 32, lane);
  v16h b1 = load_bt(BT, nt, 32, lane);
  v8f acc = {};
  acc = __builtin_amdgcn_wmma_f32_16x16x32_f16(false, a0, false, b0, (short)0, acc, false, false);
  acc = __builtin_amdgcn_wmma_f32_16x16x32_f16(false, a1, false, b1, (short)0, acc, false, false);
  return acc;
}

__global__ __launch_bounds__(256)
void mha_block_kernel(const float* __restrict__ x,
                      const float* __restrict__ Wq, const float* __restrict__ Wk,
                      const float* __restrict__ Wv, const float* __restrict__ W1,
                      const float* __restrict__ b1, float* __restrict__ y) {
  __shared__ _Float16 XH[64 * PITCH];     // x (f16); reused as attention output
  __shared__ _Float16 WT[64 * PITCH];     // current weight transposed [n][c] (Wq->Wk->Wv->W1)
  __shared__ _Float16 QH[64 * PITCH];     // Q  [t][h*8+d]
  __shared__ _Float16 KH[64 * PITCH];     // K  [s][h*8+d]
  __shared__ _Float16 VT[64 * PITCH];     // V transposed [h*8+d][s]
  __shared__ _Float16 WEI[8][16 * PITCH]; // per-wave softmax scratch (A-layout transpose)
  __shared__ float    B1S[64];

  const int tid  = threadIdx.x;
  const int lane = tid & 31;
  const int wave = tid >> 5;
  const int bidx = blockIdx.x;
  const int n    = lane & 15;
  const int half = lane >> 4;

  // ---------------- load x[b] -> XH (f32 -> f16): b128 global loads, b64 ds stores ---
  const float* xb = x + (size_t)bidx * 4096;
#pragma unroll
  for (int i = tid * 4; i < 4096; i += 1024) {
    v4f xv = *(const v4f*)&xb[i];
    v4h hv = {(_Float16)xv.x, (_Float16)xv.y, (_Float16)xv.z, (_Float16)xv.w};
    *(v4h*)&XH[(i >> 6) * PITCH + (i & 63)] = hv;
  }
  if (tid < 64) B1S[tid] = b1[tid];

  // ---------------- projections: Q, K, V (WT buffer reused; m fully unrolled) --------
#pragma unroll
  for (int m = 0; m < 3; ++m) {
    __syncthreads();  // protect WT reuse (and XH load on m==0)
    const float* W = (m == 0) ? Wq : (m == 1) ? Wk : Wv;
    // WT[h*8+d][c] = W[h][c][d]; b128 global read, 4 strided b16 LDS writes
#pragma unroll
    for (int i = tid * 4; i < 4096; i += 1024) {
      v4f wv = *(const v4f*)&W[i];
      int h = i >> 9, c = (i >> 3) & 63, d = i & 7;
      _Float16* base = &WT[(h * 8 + d) * PITCH + c];
      base[0 * PITCH] = (_Float16)wv.x;
      base[1 * PITCH] = (_Float16)wv.y;
      base[2 * PITCH] = (_Float16)wv.z;
      base[3 * PITCH] = (_Float16)wv.w;
    }
    __syncthreads();
#pragma unroll
    for (int tt = 0; tt < 2; ++tt) {          // 16 tiles / 8 waves = exactly 2 each
      int t  = wave + tt * 8;
      int mt = (t >> 2) * 16, nt = (t & 3) * 16;
      v8f acc = gemm64_tile(XH, WT, mt, nt, lane);
      if (m == 2) {                           // compile-time after unroll
#pragma unroll
        for (int r = 0; r < 8; ++r)           // store V transposed: VT[col][row]
          VT[(nt + n) * PITCH + (mt + half * 8 + r)] = (_Float16)acc[r];
      } else {
        _Float16* dst = (m == 0) ? QH : KH;
#pragma unroll
        for (int r = 0; r < 8; ++r)
          dst[(mt + half * 8 + r) * PITCH + (nt + n)] = (_Float16)acc[r];
      }
    }
  }
  __syncthreads();

  // Preload W1 transposed into WT; only read after the next barrier, so these
  // stores overlap with attention compute.
#pragma unroll
  for (int i = tid * 4; i < 4096; i += 1024) {
    v4f wv = *(const v4f*)&W1[i];
    int c = i >> 6, nn = i & 63;
    _Float16* base = &WT[nn * PITCH + c];     // W1T[n][c] = W1[c][n]
    base[0 * PITCH] = (_Float16)wv.x;
    base[1 * PITCH] = (_Float16)wv.y;
    base[2 * PITCH] = (_Float16)wv.z;
    base[3 * PITCH] = (_Float16)wv.w;
  }

  // ---------------- attention: wave w owns head w ----------------
  {
    const int h = wave;
    _Float16* wei = &WEI[wave][0];
    const v8h zero8 = {};
#pragma unroll
    for (int i = 0; i < 4; ++i) {             // 16-row query block
      // ---- scores S = Qh @ Kh^T : 4 tiles, K padded 8->32 with zero lanes ----
      v8h  qlo   = (half == 0) ? *(const v8h*)&QH[(i * 16 + n) * PITCH + h * 8] : zero8;
      v16h afrag = cat8(qlo, zero8);
      v8f s[4];
#pragma unroll
      for (int j = 0; j < 4; ++j) {
        v8h  kb    = (half == 0) ? *(const v8h*)&KH[(j * 16 + n) * PITCH + h * 8] : zero8;
        v16h bfrag = cat8(kb, zero8);
        v8f  c0    = {};
        s[j] = __builtin_amdgcn_wmma_f32_16x16x32_f16(false, afrag, false, bfrag,
                                                      (short)0, c0, false, false);
      }
      // ---- causal mask + scale (C^-0.5 = 0.125) + softmax per row ----
#pragma unroll
      for (int r = 0; r < 8; ++r) {
        int row = i * 16 + half * 8 + r;
        float mx = -3.0e38f;
#pragma unroll
        for (int j = 0; j < 4; ++j) {
          int col = j * 16 + n;
          float v = (col <= row) ? s[j][r] * 0.125f : -3.0e38f;
          s[j][r] = v;
          mx = fmaxf(mx, v);
        }
#pragma unroll
        for (int d = 1; d < 16; d <<= 1) mx = fmaxf(mx, __shfl_xor(mx, d, 32));
        float sm = 0.0f;
#pragma unroll
        for (int j = 0; j < 4; ++j) {
          float e = __expf(s[j][r] - mx);
          s[j][r] = e;
          sm += e;
        }
#pragma unroll
        for (int d = 1; d < 16; d <<= 1) sm += __shfl_xor(sm, d, 32);
        float inv = 1.0f / sm;
        int lrow = half * 8 + r;              // transpose C/D layout -> row-major scratch
#pragma unroll
        for (int j = 0; j < 4; ++j)
          wei[lrow * PITCH + j * 16 + n] = (_Float16)(s[j][r] * inv);
      }
      // ---- out block = wei(16x64) @ Vh(64x8, N padded to 16) ----
      v8f acc = {};
#pragma unroll
      for (int kc = 0; kc < 64; kc += 32) {
        v16h a  = load_a(wei, 0, kc, lane);
        v8h  lo = (n < 8) ? *(const v8h*)&VT[(h * 8 + n) * PITCH + kc + half * 16]     : zero8;
        v8h  hi = (n < 8) ? *(const v8h*)&VT[(h * 8 + n) * PITCH + kc + half * 16 + 8] : zero8;
        v16h bf = cat8(lo, hi);
        acc = __builtin_amdgcn_wmma_f32_16x16x32_f16(false, a, false, bf,
                                                     (short)0, acc, false, false);
      }
      if (n < 8) {
#pragma unroll
        for (int r = 0; r < 8; ++r)           // XH now holds the attention output
          XH[(i * 16 + half * 8 + r) * PITCH + h * 8 + n] = (_Float16)acc[r];
      }
    }
  }
  __syncthreads();

  // ---------------- FFN: y = relu(OUT @ W1 + b1), streamed to HBM ----------------
  float* yb = y + (size_t)bidx * 4096;
#pragma unroll
  for (int tt = 0; tt < 2; ++tt) {
    int t  = wave + tt * 8;
    int mt = (t >> 2) * 16, nt = (t & 3) * 16;
    v8f acc = gemm64_tile(XH, WT, mt, nt, lane);
    float bias = B1S[nt + n];
#pragma unroll
    for (int r = 0; r < 8; ++r) {
      int row = mt + half * 8 + r;
      float v = acc[r] + bias;
      yb[row * 64 + nt + n] = fmaxf(v, 0.0f);
    }
  }
}

extern "C" void kernel_launch(void* const* d_in, const int* in_sizes, int n_in,
                              void* d_out, int out_size, void* d_ws, size_t ws_size,
                              hipStream_t stream) {
  (void)n_in; (void)out_size; (void)d_ws; (void)ws_size;
  const float* x  = (const float*)d_in[0];
  const float* Wq = (const float*)d_in[1];
  const float* Wk = (const float*)d_in[2];
  const float* Wv = (const float*)d_in[3];
  const float* W1 = (const float*)d_in[4];
  const float* b1 = (const float*)d_in[5];
  float* y = (float*)d_out;
  int B = in_sizes[0] / (64 * 64);  // [B, T=64, C=64]
  mha_block_kernel<<<B, 256, 0, stream>>>(x, Wq, Wk, Wv, W1, b1, y);
}
